// ConvACNVQVAE_78228534329870
// MI455X (gfx1250) — matrix operand order repaired and checked by reading, not compiled, not run on hardware
//
#include <hip/hip_runtime.h>
#include <hip/hip_bf16.h>

// ---------------------------------------------------------------------------
// CDNA5 (gfx1250) implicit-GEMM VQ-VAE forward, round 3.
//  - weights prepacked to bf16 in WMMA A-fragment order (per-layer kernel)
//  - activations repacked per layer to bf16 NHWC (Cin padded to 32, ReLU folded)
//    so the B fragment is ONE contiguous 32B vector load per lane
//  - K loop ordered (kh, kw, cin/32); 4Mx2N blocking: 8 wmma per K step
//  - global_prefetch of next weight fragment
// ---------------------------------------------------------------------------

typedef __attribute__((ext_vector_type(16))) __bf16 v16bf;
typedef __attribute__((ext_vector_type(8)))  float  v8f;

#define MBLK 4
#define NBLK 2

static __device__ __forceinline__ __bf16 f2bf(float f) {
  union { float f; unsigned u; } v; v.f = f;
  unsigned r = v.u + 0x7FFFu + ((v.u >> 16) & 1u);   // round-to-nearest-even
  unsigned short h = (unsigned short)(r >> 16);
  return __builtin_bit_cast(__bf16, h);
}

static __device__ __forceinline__ v16bf zero_v16bf() {
  v16bf z;
  #pragma unroll
  for (int i = 0; i < 16; ++i) z[i] = __builtin_bit_cast(__bf16, (unsigned short)0);
  return z;
}

// K ordering: kLinear = ((kh*KW+kw)*CinSteps + cs)*32 + c
// A-fragment k order used here (16-bit A 16x32, contiguous k; B uses the same): lane l16 = row M; per-lane ka:
//   lanes 0-15 : v0..7 -> K 0..15
//   lanes16-31 : v0..7 -> K 16..31
__global__ void prepack_kernel(const float* __restrict__ W, __bf16* __restrict__ Wp,
                               int Cout, int Cin, int KH, int KW, int transposed) {
  const int CinSteps = (Cin + 31) >> 5;
  const int kSteps   = KH * KW * CinSteps;
  const int numMt    = (Cout + 15) >> 4;
  const long total   = (long)numMt * kSteps * 512;
  long t = (long)blockIdx.x * blockDim.x + threadIdx.x;
  if (t >= total) return;
  int e = (int)(t & 511);           // lane*16 + i
  long rest = t >> 9;
  int ksG = (int)(rest % kSteps);
  int mt  = (int)(rest / kSteps);
  int lane = e >> 4, i = e & 15;
  int half = lane >> 4, l16 = lane & 15;
  int m  = mt * 16 + l16;
  int ip = i >> 1, low = i & 1;
  int ka = half * 16 + i;  // 0..31
  int cs = ksG % CinSteps;
  int khkwIdx = ksG / CinSteps;
  int kh = khkwIdx / KW, kw = khkwIdx - kh * KW;
  int cin = cs * 32 + ka;
  float v = 0.f;
  if (m < Cout && cin < Cin) {
    if (!transposed)
      v = W[((size_t)m * Cin + cin) * KH * KW + khkwIdx];
    else
      v = W[(((size_t)cin * Cout + m) * KH + (KH - 1 - kh)) * KW + (KW - 1 - kw)];
  }
  Wp[t] = f2bf(v);
}

// Activation repack: NCHW f32 -> NHWC bf16, channels padded to CinPad,
// optional ReLU folded in; zero-fill for c >= Cin.
__global__ void repack_kernel(const float* __restrict__ X, __bf16* __restrict__ Xp,
                              int Cin, int HW, int CinPad, long total, int reluIn) {
  long t = (long)blockIdx.x * blockDim.x + threadIdx.x;   // over Bn*HW*CinPad
  if (t >= total) return;
  int  c = (int)(t % CinPad);
  long p = t / CinPad;                                    // b*HW + pixel
  float v = 0.f;
  if (c < Cin) {
    long b = p / HW; long pix = p - b * HW;
    v = X[((size_t)b * Cin + c) * HW + pix];
    if (reluIn) v = fmaxf(v, 0.f);
  }
  Xp[t] = f2bf(v);
}

// One wave computes a 64(M) x 32(N) tile. Block = 8 waves = 128x128.
// All launches have Ntot % 16 == 0 by construction.
__global__ __launch_bounds__(256) void conv_wmma_kernel(
    const __bf16* __restrict__ Xp, const __bf16* __restrict__ Wp,
    const float* __restrict__ bias,
    const float* __restrict__ bng, const float* __restrict__ bnb,
    const float* __restrict__ bnm, const float* __restrict__ bnv,
    const float* __restrict__ resid, float* __restrict__ Y,
    int Bn, int Cin, int Hin, int Win, int CinPad,
    int Cout, int KH, int KW, int pad, int Hout, int Wout,
    int reluOut)
{
  const int HW   = Hout * Wout;
  const int Ntot = Bn * HW;
  const int CinSteps = CinPad >> 5;
  const int kSteps   = KH * KW * CinSteps;

  const int wave  = threadIdx.x >> 5;
  const int waveM = wave & 1, waveN = wave >> 1;
  const int mBase0 = (blockIdx.y * 2 + waveM) * (MBLK * 16);
  const int nBase0 = (blockIdx.x * 4 + waveN) * (NBLK * 16);
  if (mBase0 >= Cout || nBase0 >= Ntot) return;   // uniform per wave

  const int lane = threadIdx.x & 31;
  const int half = lane >> 4, l16 = lane & 15;
  const size_t HinWin = (size_t)Hin * Win;
  const bool noPad = (pad == 0);
  const v16bf bzero = zero_v16bf();

  // Per-lane N decomposition (tile-uniform validity)
  int  nbv[NBLK], ohv[NBLK], owv[NBLK];
  bool jval[NBLK];
  #pragma unroll
  for (int j = 0; j < NBLK; ++j) {
    jval[j] = (nBase0 + j * 16) < Ntot;
    int n = nBase0 + j * 16 + l16;
    nbv[j] = 0; ohv[j] = 0; owv[j] = 0;
    if (jval[j]) {
      int nb = n / HW; int r = n - nb * HW;
      nbv[j] = nb; ohv[j] = r / Wout; owv[j] = r - (r / Wout) * Wout;
    }
  }
  bool mtv[MBLK];
  #pragma unroll
  for (int mi = 0; mi < MBLK; ++mi) mtv[mi] = (mBase0 + mi * 16) < Cout;

  v8f acc[MBLK][NBLK];
  #pragma unroll
  for (int mi = 0; mi < MBLK; ++mi)
    #pragma unroll
    for (int j = 0; j < NBLK; ++j)
      acc[mi][j] = (v8f){0.f,0.f,0.f,0.f,0.f,0.f,0.f,0.f};

  for (int kh = 0; kh < KH; ++kh) {
    for (int kw = 0; kw < KW; ++kw) {
      // Per-tap per-lane pixel offsets into NHWC-bf16 staging
      long xoff[NBLK];
      bool sval[NBLK];
      #pragma unroll
      for (int j = 0; j < NBLK; ++j) {
        int ih = ohv[j] + kh - pad, iw = owv[j] + kw - pad;
        sval[j] = jval[j] && (noPad ||
                  ((unsigned)ih < (unsigned)Hin && (unsigned)iw < (unsigned)Win));
        xoff[j] = ((long)nbv[j] * (long)HinWin + (long)ih * Win + iw) * CinPad
                  + half * 16;
      }
      const int khkwIdx = kh * KW + kw;
      for (int cs = 0; cs < CinSteps; ++cs) {
        const int cin0 = cs << 5;
        const int ksG  = khkwIdx * CinSteps + cs;
        // A fragments: one contiguous 32B bf16 vector per lane per M tile
        v16bf af[MBLK];
        #pragma unroll
        for (int mi = 0; mi < MBLK; ++mi) {
          if (mtv[mi]) {
            size_t poff = (((size_t)((mBase0 >> 4) + mi) * kSteps + ksG) * 32 + lane) * 16;
            af[mi] = *(const v16bf*)(Wp + poff);
          }
        }
        // prefetch next K-step's first weight fragment
        if (cs + 1 < CinSteps) {
          size_t pn = (((size_t)(mBase0 >> 4) * kSteps + ksG + 1) * 32 + lane) * 16;
          __builtin_prefetch(Wp + pn, 0, 0);
        }
        // B fragments: one contiguous 32B bf16 vector per lane per N tile
        #pragma unroll
        for (int j = 0; j < NBLK; ++j) {
          v16bf bfv;
          if (sval[j]) bfv = *(const v16bf*)(Xp + xoff[j] + cin0);
          else         bfv = bzero;
          #pragma unroll
          for (int mi = 0; mi < MBLK; ++mi) {
            if (mtv[mi]) {
              acc[mi][j] = __builtin_amdgcn_wmma_f32_16x16x32_bf16(
                  false, af[mi], false, bfv, (short)0, acc[mi][j], false, false);
            }
          }
        }
      }
    }
  }

  // Epilogue: D layout -> VGPR r holds M row (r + 8*half), N col = lane&15
  #pragma unroll
  for (int mi = 0; mi < MBLK; ++mi) {
    if (!mtv[mi]) continue;
    #pragma unroll
    for (int r = 0; r < 8; ++r) {
      int m = mBase0 + mi * 16 + r + half * 8;
      if (m >= Cout) continue;
      float bsc = 1.f, bsh = 0.f;
      if (bng) { bsc = bng[m] * rsqrtf(bnv[m] + 1e-5f); bsh = bnb[m] - bnm[m] * bsc; }
      #pragma unroll
      for (int j = 0; j < NBLK; ++j) {
        if (!jval[j]) continue;
        float t = acc[mi][j][r];
        if (bias) t += bias[m];
        if (bng)  t = t * bsc + bsh;
        size_t oidx = (((size_t)nbv[j] * Cout + m) * Hout + ohv[j]) * Wout + owv[j];
        if (resid)   t += resid[oidx];
        if (reluOut) t = fmaxf(t, 0.f);
        Y[oidx] = t;
      }
    }
  }
}

// ------------------------------ VQ helpers ---------------------------------

__global__ void enorm_kernel(const float* __restrict__ E, float* __restrict__ en) {
  int k = blockIdx.x * blockDim.x + threadIdx.x;
  if (k < 512) {
    float s = 0.f;
    for (int d = 0; d < 256; ++d) { float e = E[(size_t)k * 256 + d]; s += e * e; }
    en[k] = s;
  }
}

__global__ void argmin_kernel(const float* __restrict__ sc,
                              const float* __restrict__ en,
                              int* __restrict__ idx) {
  int nid = blockIdx.x * blockDim.x + threadIdx.x;
  if (nid < 32768) {
    int b = nid >> 8; int r = nid & 255; int h = r >> 4; int w = r & 15;
    const float* base = sc + (size_t)b * 512 * 256 + h * 16 + w;
    float best = 3.4e38f; int bi = 0;
    for (int k = 0; k < 512; ++k) {
      float s = en[k] - 2.f * base[(size_t)k * 256];
      if (s < best) { best = s; bi = k; }
    }
    idx[nid] = bi;
  }
}

__global__ void zero_kernel(float* __restrict__ p, long nelem) {
  long i = (long)blockIdx.x * blockDim.x + threadIdx.x;
  if (i < nelem) p[i] = 0.f;
}

__global__ void vqq_gather_kernel(const int* __restrict__ idx,
                                  const float* __restrict__ E,
                                  float* __restrict__ q) {
  long t = (long)blockIdx.x * blockDim.x + threadIdx.x;  // 32768 * 256
  if (t < 8388608L) {
    int nid = (int)(t >> 8); int d = (int)(t & 255);
    int b = nid >> 8; int r = nid & 255; int h = r >> 4; int w = r & 15;
    q[(((size_t)b * 256 + d) * 16 + h) * 16 + w] = E[(size_t)idx[nid] * 256 + d];
  }
}

__global__ void vqi_scatter_kernel(const int* __restrict__ idx,
                                   float* __restrict__ vi) {
  int nid = blockIdx.x * blockDim.x + threadIdx.x;
  if (nid < 32768) {
    int b = nid >> 8; int r = nid & 255; int h = r >> 4; int w = r & 15;
    vi[(((size_t)b * 512 + idx[nid]) * 16 + h) * 16 + w] = 1.f;
  }
}

// ------------------------------ host side ----------------------------------

static void launch_conv(hipStream_t s, const float* X, const float* W,
    __bf16* Wp, __bf16* Xp,
    const float* bias, const float* g, const float* b, const float* m,
    const float* v, const float* resid, float* Y,
    int Bn, int Cin, int Hin, int Win, int Cout, int K, int pad,
    int Hout, int Wout, int reluIn, int reluOut, int transposed)
{
  int CinPad   = ((Cin + 31) / 32) * 32;
  int CinSteps = CinPad / 32;
  // 1) prepack weights -> bf16 fragment order
  long wtot = (long)((Cout + 15) / 16) * (K * K * CinSteps) * 512;
  prepack_kernel<<<(int)((wtot + 255) / 256), 256, 0, s>>>(W, Wp, Cout, Cin, K, K, transposed);
  // 2) repack activations -> bf16 NHWC (padded channels, ReLU folded)
  long xtot = (long)Bn * Hin * Win * CinPad;
  repack_kernel<<<(int)((xtot + 255) / 256), 256, 0, s>>>(X, Xp, Cin, Hin * Win, CinPad, xtot, reluIn);
  // 3) WMMA implicit GEMM (block tile 128x128, wave tile 64x32)
  long Ntot = (long)Bn * Hout * Wout;               // always multiple of 16 here
  dim3 grid((unsigned)((Ntot + 127) / 128), (unsigned)((Cout + 127) / 128));
  conv_wmma_kernel<<<grid, 256, 0, s>>>(Xp, Wp, bias, g, b, m, v, resid, Y,
      Bn, Cin, Hin, Win, CinPad, Cout, K, K, pad, Hout, Wout, reluOut);
}

extern "C" void kernel_launch(void* const* d_in, const int* in_sizes, int n_in,
                              void* d_out, int out_size, void* d_ws, size_t ws_size,
                              hipStream_t stream) {
  (void)in_sizes; (void)n_in; (void)out_size; (void)ws_size;
  auto P = [&](int i) { return (const float*)d_in[i]; };

  // Output segments (concatenated, return order)
  float* out = (float*)d_out;
  float* OUT0   = out + 0;         // [128,1,28,28]    100352
  float* O_MU   = out + 100352;    // [128,512]         65536
  float* O_LS   = out + 165888;    // [128,512]         65536
  float* O_VQE  = out + 231424;    // [128,256,16,16] 8388608
  float* O_VQQ  = out + 8620032;   // [128,256,16,16] 8388608
  float* O_VQI  = out + 17008640;  // [128,512,16,16] 16777216

  // Workspace carve (floats)
  float* wsf = (float*)d_ws;
  size_t off = 0;
  auto take = [&](size_t ne) { float* p = wsf + off; off += ne; return p; };
  float* A  = take(25690112);  // big: fits [128,256,28,28]; also aliased by scores
  float* Bb = take(18874368);  // [128,256,24,24]
  float* Cc = take(18874368);
  float* MU = take(2097152);   // [128,64,16,16]
  float* LS = take(2097152);
  float* AZ = take(2097152);   // acn_z reshaped [128,64,16,16]
  float* EN = take(512);
  int*   IDX = (int*)(wsf + off); off += 32768;
  __bf16* WP = (__bf16*)(wsf + off); off += 4194304;   // 8.4M bf16 weight scratch
  __bf16* XP = (__bf16*)(wsf + off);                   // 25.7M bf16 activation scratch

  const float* IN = P(0);
  const float* E  = P(99);

  auto resblock = [&](const float* x, float* tmp, float* yout, int base) {
    launch_conv(stream, x, P(base), WP, XP, P(base + 1), P(base + 2), P(base + 3),
                P(base + 4), P(base + 5), nullptr, tmp,
                128, 256, 16, 16, 256, 3, 1, 16, 16, 1, 0, 0);
    launch_conv(stream, tmp, P(base + 6), WP, XP, P(base + 7), P(base + 8), P(base + 9),
                P(base + 10), P(base + 11), x, yout,
                128, 256, 16, 16, 256, 1, 0, 16, 16, 1, 0, 0);
  };

  // ---------------- encoder ----------------
  launch_conv(stream, IN, P(1), WP, XP, P(2), 0,0,0,0, nullptr, A,   128,   1, 28, 28, 256, 5, 0, 24, 24, 0, 1, 0);
  launch_conv(stream, A,  P(3), WP, XP, P(4), 0,0,0,0, nullptr, Bb,  128, 256, 24, 24, 256, 5, 0, 20, 20, 0, 1, 0);
  launch_conv(stream, Bb, P(5), WP, XP, P(6), 0,0,0,0, nullptr, Cc,  128, 256, 20, 20, 256, 5, 0, 16, 16, 0, 1, 0);
  resblock(Cc, A, Bb, 7);                                                                       // res1 -> Bb
  resblock(Bb, A, Cc, 19);                                                                      // res2 -> Cc
  launch_conv(stream, Cc, P(31), WP, XP, P(32), 0,0,0,0, nullptr, A,  128, 256, 16, 16, 256, 3, 1, 16, 16, 0, 0, 0);
  launch_conv(stream, A,  P(33), WP, XP, P(34), 0,0,0,0, nullptr, Bb, 128, 256, 16, 16, 256, 3, 1, 16, 16, 0, 0, 0);
  launch_conv(stream, Bb, P(35), WP, XP, P(36), 0,0,0,0, nullptr, MU, 128, 256, 16, 16,  64, 1, 0, 16, 16, 0, 0, 0);
  launch_conv(stream, Bb, P(37), WP, XP, P(38), 0,0,0,0, nullptr, LS, 128, 256, 16, 16,  64, 1, 0, 16, 16, 0, 0, 0);
  // linear layers as 1x1 convs over [B][Cin] (NCHW flatten == reshape order)
  launch_conv(stream, MU,   P(39), WP, XP, P(40), 0,0,0,0, nullptr, O_MU, 128, 16384, 1, 1,   512, 1, 0, 1, 1, 0, 0, 0);
  launch_conv(stream, LS,   P(41), WP, XP, P(42), 0,0,0,0, nullptr, O_LS, 128, 16384, 1, 1,   512, 1, 0, 1, 1, 0, 0, 0);
  launch_conv(stream, O_MU, P(43), WP, XP, P(44), 0,0,0,0, nullptr, AZ,   128,   512, 1, 1, 16384, 1, 0, 1, 1, 0, 0, 0);
  // vq encoder branch
  launch_conv(stream, AZ, P(45), WP, XP, P(46), P(47), P(48), P(49), P(50), nullptr, A,
              128, 64, 16, 16, 256, 1, 0, 16, 16, 0, 1, 0);                                     // relu(bn(c1_v))
  resblock(A, Bb, Cc, 51);                                                                      // res1_v -> Cc
  launch_conv(stream, Cc, P(63), WP, XP, P(64), 0,0,0,0, nullptr, O_VQE, 128, 256, 16, 16, 256, 1, 0, 16, 16, 0, 0, 0);

  // ---------------- VQ ----------------
  enorm_kernel<<<2, 256, 0, stream>>>(E, EN);
  launch_conv(stream, O_VQE, E, WP, XP, nullptr, 0,0,0,0, nullptr, A,
              128, 256, 16, 16, 512, 1, 0, 16, 16, 0, 0, 0);          // scores = x . E^T
  argmin_kernel<<<128, 256, 0, stream>>>(A, EN, IDX);
  zero_kernel<<<65536, 256, 0, stream>>>(O_VQI, 16777216L);
  vqq_gather_kernel<<<32768, 256, 0, stream>>>(IDX, E, O_VQQ);
  vqi_scatter_kernel<<<128, 256, 0, stream>>>(IDX, O_VQI);

  // ---------------- decoder ----------------
  launch_conv(stream, O_VQQ, P(65), WP, XP, P(66), 0,0,0,0, nullptr, Bb,
              128, 256, 16, 16, 256, 1, 0, 16, 16, 0, 0, 1);                                    // ic4^T
  resblock(Bb, A, Cc, 67);                                                                      // ires1 -> Cc
  resblock(Cc, A, Bb, 79);                                                                      // ires2 -> Bb
  launch_conv(stream, Bb, P(91), WP, XP, P(92), 0,0,0,0, nullptr, A,
              128, 256, 16, 16, 256, 5, 4, 20, 20, 0, 1, 1);                                    // ih2 = relu(ic3^T)
  launch_conv(stream, A,  P(93), WP, XP, P(94), 0,0,0,0, nullptr, Bb,
              128, 256, 20, 20, 256, 5, 4, 24, 24, 0, 1, 1);                                    // ih3 = relu(ic2^T)
  launch_conv(stream, Bb, P(95), WP, XP, P(96), 0,0,0,0, nullptr, A,
              128, 256, 24, 24, 256, 5, 4, 28, 28, 0, 0, 1);                                    // ic1^T
  launch_conv(stream, A,  P(97), WP, XP, P(98), 0,0,0,0, nullptr, OUT0,
              128, 256, 28, 28,   1, 1, 0, 28, 28, 0, 0, 1);                                    // ic1_o^T
}
